// Head_72808285602304
// MI455X (gfx1250) — matrix operand order repaired and checked by reading, not compiled, not run on hardware
//
#include <hip/hip_runtime.h>

// ---------------------------------------------------------------------------
// Single-head causal attention for MI455X (gfx1250, wave32, WMMA).
//   B=512, T=256, C=384, H=64.  Inputs fp32; q/k/v staged as f16 in d_ws,
//   all matmuls via v_wmma_f32_16x16x32_f16 (f32 accumulate).
//   K staged into LDS via GLOBAL_LOAD_ASYNC_TO_LDS (ASYNCcnt) when available.
// ---------------------------------------------------------------------------

typedef __attribute__((ext_vector_type(16))) _Float16 v16h;
typedef __attribute__((ext_vector_type(8)))  float    v8f;
typedef __attribute__((ext_vector_type(4)))  int      v4i;

#define B_ 512
#define T_ 256
#define C_ 384
#define H_ 64
#define SCALE 0.05103103630798287f   /* 384^-0.5 (reference scales by C^-0.5) */

#if defined(__has_builtin)
#  if __has_builtin(__builtin_amdgcn_global_load_async_to_lds_b128)
#    define HAS_ASYNC_LDS 1
#  endif
#endif

static __device__ __forceinline__ v8f wmma_f16(v16h a, v16h b, v8f c) {
  // D = A(16x32 f16) * B(32x16 f16) + C(16x16 f32)
  return __builtin_amdgcn_wmma_f32_16x16x32_f16(
      /*neg_a=*/false, a, /*neg_b=*/false, b,
      /*c_mod=*/(short)0, c, /*reuse_a=*/false, /*reuse_b=*/false);
}

static __device__ __forceinline__ void wait_async0() {
#if defined(__has_builtin) && __has_builtin(__builtin_amdgcn_s_wait_asynccnt)
  __builtin_amdgcn_s_wait_asynccnt(0);
#else
  asm volatile("s_wait_asynccnt 0x0" ::: "memory");
#endif
}

// ---------------------------------------------------------------------------
// Kernel 0: repack Wq|Wk|Wv (f32, [C,H]) into f16 B-matrix lane layout.
//   Wh[((t*12 + ks)*32 + lane)*16 + e] = W_t[(ks*32 + (lane>>4)*16 + e)*H + (t&3)*16 + (lane&15)]
//   so kernel 1's B loads are contiguous 32B runs per lane.
// ---------------------------------------------------------------------------
__global__ __launch_bounds__(256) void packw_kernel(
    const float* __restrict__ Wq, const float* __restrict__ Wk,
    const float* __restrict__ Wv, _Float16* __restrict__ Wh) {
  const int idx = blockIdx.x * 256 + threadIdx.x;      // 0 .. 3*12*32*16-1
  const int e    = idx & 15;
  const int lane = (idx >> 4) & 31;
  const int ks   = (idx >> 9) % 12;
  const int t    = idx / (16 * 32 * 12);               // 0..11 (which*4 + nt)
  const int half = lane >> 4, mm = lane & 15;
  const int which = t >> 2, nt = t & 3;
  const int kk  = ks * 32 + half * 16 + e;
  const int col = nt * 16 + mm;
  const float* W = (which == 0) ? Wq : (which == 1) ? Wk : Wv;
  Wh[idx] = (_Float16)W[(size_t)kk * H_ + col];
}

// ---------------------------------------------------------------------------
// Kernel 1: q,k,v = x @ {Wq,Wk,Wv}.  One wave per 16-row tile of x
// computes all 12 output 16x16 tiles -> x is read exactly once (201 MB).
// B tiles batched 6-at-a-time into registers so loads clause up and the
// 6 WMMAs issue back-to-back instead of load/wait/wmma lockstep.
// ---------------------------------------------------------------------------
__global__ __launch_bounds__(256) void proj_kernel(
    const float* __restrict__ x, const _Float16* __restrict__ Wh,
    _Float16* __restrict__ qo, _Float16* __restrict__ ko,
    _Float16* __restrict__ vo) {
  const int rowtile = blockIdx.x * 8 + (threadIdx.x >> 5);  // 0..8191
  const int lane = threadIdx.x & 31;
  const int m = lane & 15, half = lane >> 4;

  const float* xr = x + (size_t)(rowtile * 16 + m) * C_;

  v8f acc[12] = {};
  #pragma unroll
  for (int ks = 0; ks < 12; ++ks) {
    const int k0 = ks * 32;
    // A tile (16x32 f16): lane holds row m; K = half*16 + e (contiguous, matches packed B).
    v16h a;
    #pragma unroll
    for (int e = 0; e < 8; ++e) {
      a[e]     = (_Float16)xr[k0 + half * 16 + e];
      a[8 + e] = (_Float16)xr[k0 + half * 16 + 8 + e];
    }
    __builtin_prefetch(xr + k0 + 128, 0, 3);   // global_prefetch_b8 ahead
    #pragma unroll
    for (int g = 0; g < 2; ++g) {
      v16h bt[6];
      #pragma unroll
      for (int u = 0; u < 6; ++u) {
        const int t = g * 6 + u;
        const _Float16* bh = Wh + ((size_t)(t * 12 + ks) * 32 + lane) * 16;
        #pragma unroll
        for (int e = 0; e < 16; ++e) bt[u][e] = bh[e];
      }
      #pragma unroll
      for (int u = 0; u < 6; ++u)
        acc[g * 6 + u] = wmma_f16(a, bt[u], acc[g * 6 + u]);
    }
  }
  // C layout: lanes 0-15 -> M=r, lanes 16-31 -> M=8+r; N = lane&15.
  #pragma unroll
  for (int t = 0; t < 12; ++t) {
    _Float16* O = ((t >> 2) == 0) ? qo : ((t >> 2) == 1) ? ko : vo;
    _Float16* orow = O + (size_t)(rowtile * 16 + half * 8) * H_ + (t & 3) * 16 + m;
    #pragma unroll
    for (int r = 0; r < 8; ++r) orow[(size_t)r * H_] = (_Float16)acc[t][r];
  }
}

// ---------------------------------------------------------------------------
// Kernel 2: causal attention for one batch per block (8 waves, 256 thr).
//   LDS: K row-major (32KB, async-loaded) + V transposed (32KB) +
//        per-wave P bounce (64KB).  Wave w handles query tiles {w, 15-w}.
// ---------------------------------------------------------------------------
__global__ __launch_bounds__(256) void attn_kernel(
    const _Float16* __restrict__ q, const _Float16* __restrict__ k,
    const _Float16* __restrict__ v, float* __restrict__ out) {
  __shared__ __align__(16) _Float16 kl[T_ * H_];      // [t][h]   32 KB
  __shared__ __align__(16) _Float16 vt[H_ * T_];      // [h][t]   32 KB (transposed)
  __shared__ __align__(16) _Float16 pl[8 * 16 * T_];  // per-wave P, 64 KB

  const int b = blockIdx.x;
  const int tid = threadIdx.x;
  const int wid = tid >> 5, lane = tid & 31;
  const int m = lane & 15, half = lane >> 4;

  // ---- stage K: global -> LDS, 128B per thread ----
#if defined(HAS_ASYNC_LDS)
  {
    const _Float16* gk = k + (size_t)b * T_ * H_;
    #pragma unroll
    for (int i = 0; i < 8; ++i) {
      const int idx = (tid + 256 * i) * 8;   // halves (16B chunks)
      __builtin_amdgcn_global_load_async_to_lds_b128(
          (__attribute__((address_space(1))) v4i*)(gk + idx),
          (__attribute__((address_space(3))) v4i*)(kl + idx),
          /*offset=*/0, /*cpol=*/0);
    }
  }
#else
  {
    const uint4* src = (const uint4*)(k + (size_t)b * T_ * H_);
    uint4* dst = (uint4*)kl;
    #pragma unroll
    for (int i = 0; i < 8; ++i) dst[tid + 256 * i] = src[tid + 256 * i];
  }
#endif
  // ---- stage V transposed: thread tid owns row t=tid ----
  {
    const _Float16* vr = v + ((size_t)b * T_ + tid) * H_;
    #pragma unroll
    for (int h = 0; h < H_; ++h) vt[h * T_ + tid] = vr[h];
  }
#if defined(HAS_ASYNC_LDS)
  wait_async0();                 // own async K chunks complete
#endif
  __syncthreads();               // cross-wave visibility

  _Float16* pw = pl + wid * 16 * T_;

  for (int pass = 0; pass < 2; ++pass) {
    const int qi  = pass ? (15 - wid) : wid;   // query tile index, wave-uniform
    const int nkj = qi + 1;                    // causal: k-tiles 0..qi

    // Q A-tiles for K=0..31 and K=32..63 (global, f16, contiguous runs)
    const _Float16* qr = q + ((size_t)b * T_ + qi * 16 + m) * H_;
    v16h a0, a1;
    #pragma unroll
    for (int e = 0; e < 8; ++e) {
      a0[e] = qr[half * 16 + e];       a0[8 + e] = qr[half * 16 + 8 + e];
      a1[e] = qr[32 + half * 16 + e];  a1[8 + e] = qr[40 + half * 16 + e];
    }

    // ---- S = (Q K^T) * scale, causal-masked; f32 accum stays in VGPRs ----
    float sc[16][8];
    #pragma unroll
    for (int kj = 0; kj < 16; ++kj) {
      if (kj < nkj) {                               // uniform scalar branch
        v16h b0, b1;
        const _Float16* kr = kl + (kj * 16 + m) * H_;   // m = B-matrix column n
        #pragma unroll
        for (int e = 0; e < 16; ++e) {
          b0[e] = kr[half * 16 + e];
          b1[e] = kr[32 + half * 16 + e];
        }
        v8f s = {};
        s = wmma_f16(a0, b0, s);
        s = wmma_f16(a1, b1, s);
        #pragma unroll
        for (int r = 0; r < 8; ++r) {
          float val = s[r] * SCALE;
          if (kj == qi && m > half * 8 + r) val = -1e30f;  // col>row on diagonal
          sc[kj][r] = val;
        }
      }
    }

    // ---- softmax: row max/sum via 16-lane shuffle reductions ----
    float rmax[8], rsum[8];
    #pragma unroll
    for (int r = 0; r < 8; ++r) rmax[r] = -1e30f;
    #pragma unroll
    for (int kj = 0; kj < 16; ++kj)
      if (kj < nkj) {
        #pragma unroll
        for (int r = 0; r < 8; ++r) rmax[r] = fmaxf(rmax[r], sc[kj][r]);
      }
    #pragma unroll
    for (int r = 0; r < 8; ++r) {
      #pragma unroll
      for (int off = 8; off >= 1; off >>= 1)
        rmax[r] = fmaxf(rmax[r], __shfl_xor(rmax[r], off, 16));
      rsum[r] = 0.0f;
    }
    // exp, accumulate row sums, spill P (f16) to private LDS region
    #pragma unroll
    for (int kj = 0; kj < 16; ++kj)
      if (kj < nkj) {
        #pragma unroll
        for (int r = 0; r < 8; ++r) {
          float p = __expf(sc[kj][r] - rmax[r]);
          rsum[r] += p;
          pw[(half * 8 + r) * T_ + kj * 16 + m] = (_Float16)p;
        }
      }
    if (nkj & 1) {   // zero-pad odd tile so K=32 steps read zeros
      #pragma unroll
      for (int r = 0; r < 8; ++r)
        pw[(half * 8 + r) * T_ + nkj * 16 + m] = (_Float16)0.0f;
    }
    #pragma unroll
    for (int r = 0; r < 8; ++r) {
      #pragma unroll
      for (int off = 8; off >= 1; off >>= 1)
        rsum[r] += __shfl_xor(rsum[r], off, 16);
    }

    // ---- O = P @ V (normalize at the end: softmax is linear in P) ----
    v8f o[4] = {};
    const int nks = (nkj + 1) >> 1;          // K-steps of 32 columns
    #pragma unroll
    for (int ks = 0; ks < 8; ++ks) {
      if (ks < nks) {
        v16h ap;
        const _Float16* pr = pw + m * T_ + ks * 32;   // A row m of P
        #pragma unroll
        for (int e = 0; e < 8; ++e) {
          ap[e]     = pr[half * 16 + e];
          ap[8 + e] = pr[half * 16 + 8 + e];
        }
        #pragma unroll
        for (int ntv = 0; ntv < 4; ++ntv) {
          v16h bv;
          const _Float16* vrw = vt + (ntv * 16 + m) * T_ + ks * 32; // col h
          #pragma unroll
          for (int e = 0; e < 16; ++e) bv[e] = vrw[half * 16 + e];
          o[ntv] = wmma_f16(ap, bv, o[ntv]);
        }
      }
    }

    // ---- normalize by row sum, store fp32 output ----
    float* orow = out + ((size_t)b * T_ + qi * 16 + half * 8) * H_;
    #pragma unroll
    for (int r = 0; r < 8; ++r) {
      const float inv = 1.0f / rsum[r];
      #pragma unroll
      for (int ntv = 0; ntv < 4; ++ntv)
        orow[(size_t)r * H_ + ntv * 16 + m] = o[ntv][r] * inv;
    }
  }
}

// ---------------------------------------------------------------------------
extern "C" void kernel_launch(void* const* d_in, const int* in_sizes, int n_in,
                              void* d_out, int out_size, void* d_ws, size_t ws_size,
                              hipStream_t stream) {
  const float* x  = (const float*)d_in[0];
  const float* Wq = (const float*)d_in[1];
  const float* Wk = (const float*)d_in[2];
  const float* Wv = (const float*)d_in[3];
  float* out = (float*)d_out;

  const size_t nqkv = (size_t)B_ * T_ * H_;        // 8,388,608 halves each
  _Float16* qws = (_Float16*)d_ws;
  _Float16* kws = qws + nqkv;
  _Float16* vws = kws + nqkv;
  _Float16* Wh  = vws + nqkv;                      // 73,728 halves

  packw_kernel<<<(3 * 12 * 32 * 16) / 256, 256, 0, stream>>>(Wq, Wk, Wv, Wh);
  proj_kernel<<<(B_ * T_ / 16) / 8, 256, 0, stream>>>(x, Wh, qws, kws, vws);
  attn_kernel<<<B_, 256, 0, stream>>>(qws, kws, vws, out);
}
